// AxialAttention2dBlock_16544214024503
// MI455X (gfx1250) — matrix operand order repaired and checked by reading, not compiled, not run on hardware
//
#include <hip/hip_runtime.h>
#include <hip/hip_bf16.h>
#include <cstddef>
#include <cstdint>

// ---------------- CDNA5 (gfx1250, wave32) WMMA plumbing ----------------
typedef __attribute__((ext_vector_type(16))) __bf16 v16bf;
typedef __attribute__((ext_vector_type(8)))  float  v8f;

#define WMMA_BF16(A,B,C) \
  __builtin_amdgcn_wmma_f32_16x16x32_bf16(false,(A),false,(B),(short)0,(C),false,false)

// element j of the 16-element bf16 A fragment -> K index within the K=32 panel
// (16-bit A 16x32 layout: V0..3 = K 0..7 (lanes0-15)/8..15 (lanes16-31), V4..7 = +16)
__device__ __forceinline__ int amap(int j, int half) {
  return ((j & 8) << 1) + half * 8 + (j & 7);
}

#define EPS 1e-5f

// ---- async global->LDS staging (CDNA5 GLOBAL_LOAD_ASYNC_TO_LDS_B128, ASYNCcnt) ----
// Copies a 32x128 f32 panel (16KB) with 128 threads, 8 x 16B chunks per thread.
__device__ __forceinline__ void async_copy_panel(const float* __restrict__ gsrc,
                                                 float* lds_dst, int tid) {
  unsigned lbase = (unsigned)(uintptr_t)lds_dst;           // low 32 bits of the pointer = LDS address
  unsigned long long gbase = (unsigned long long)(uintptr_t)gsrc;
#pragma unroll
  for (int p = 0; p < 8; ++p) {
    int idx = p * 128 + tid;
    unsigned l = lbase + idx * 16;
    unsigned long long g = gbase + (unsigned long long)idx * 16;
    asm volatile("global_load_async_to_lds_b128 %0, %1, off"
                 :: "v"(l), "v"(g) : "memory");
  }
}

__device__ __forceinline__ void wait_async0() {
  asm volatile("s_wait_asynccnt 0x0" ::: "memory");
}

// ---------------- small helpers ----------------
__global__ void k_zero(float* p, int n) {
  int i = blockIdx.x * blockDim.x + threadIdx.x;
  if (i < n) p[i] = 0.f;
}

// sb = lin @ latent   (rows x L)
__global__ void k_sb(const float* __restrict__ lin, const float* __restrict__ lat,
                     float* __restrict__ sb, int rows, int L) {
  int r = blockIdx.x * blockDim.x + threadIdx.x;
  if (r >= rows) return;
  float s = 0.f;
  for (int i = 0; i < L; ++i) s += lin[r * L + i] * lat[i];
  sb[r] = s;
}

__device__ __forceinline__ void block_reduce2(float& s, float& q) {
  __shared__ float ss[256], sq[256];
  int t = threadIdx.x;
  ss[t] = s; sq[t] = q; __syncthreads();
  for (int off = blockDim.x >> 1; off > 0; off >>= 1) {
    if (t < off) { ss[t] += ss[t + off]; sq[t] += sq[t + off]; }
    __syncthreads();
  }
  s = ss[0]; q = sq[0];
}

// per-channel sum/sumsq over channel-major [C][N]
__global__ void k_stats(const float* __restrict__ x, float* __restrict__ acc, int C, int N) {
  int c = blockIdx.y;
  float s = 0.f, q = 0.f;
  for (int i = blockIdx.x * blockDim.x + threadIdx.x; i < N; i += gridDim.x * blockDim.x) {
    float v = x[(size_t)c * N + i]; s += v; q += v * v;
  }
  block_reduce2(s, q);
  if (threadIdx.x == 0) { atomicAdd(&acc[c], s); atomicAdd(&acc[C + c], q); }
}

// stats per g over buffers laid out [n=512][g=4][16384]
__global__ void k_stats_ng(const float* __restrict__ buf, float* __restrict__ acc, int chBase) {
  int g = blockIdx.y;
  float s = 0.f, q = 0.f;
  for (long e = blockIdx.x * (long)blockDim.x + threadIdx.x; e < 512L * 16384;
       e += (long)gridDim.x * blockDim.x) {
    long n = e >> 14; int ij = (int)(e & 16383);
    float v = buf[n * 65536 + g * 16384 + ij];
    s += v; q += v * v;
  }
  block_reduce2(s, q);
  if (threadIdx.x == 0) { atomicAdd(&acc[chBase + g], s); atomicAdd(&acc[12 + chBase + g], q); }
}

// stats per mid-channel over [512][64][128]
__global__ void k_stats_mid(const float* __restrict__ buf, float* __restrict__ acc) {
  int c = blockIdx.y;
  float s = 0.f, q = 0.f;
  for (int e = blockIdx.x * blockDim.x + threadIdx.x; e < 512 * 128;
       e += gridDim.x * blockDim.x) {
    int n = e >> 7, i = e & 127;
    float v = buf[(size_t)n * 8192 + c * 128 + i];
    s += v; q += v * v;
  }
  block_reduce2(s, q);
  if (threadIdx.x == 0) { atomicAdd(&acc[c], s); atomicAdd(&acc[64 + c], q); }
}

// ---------------- 1x1 conv GEMM: Y[64][N] = W[64x64] @ X ----------------
// X element (k, col): addr = (col/HW)*sB + k*sC + col%HW
__global__ void k_conv(const float* __restrict__ W, const float* __restrict__ X,
                       float* __restrict__ Y, int N, int HW, int sB, int sC) {
  const int K = 64;
  int lane = threadIdx.x & 31, wv = threadIdx.x >> 5;
  int half = lane >> 4, lm = lane & 15;
  int mt = blockIdx.y;
  int nt = blockIdx.x * 4 + wv;
  int col = nt * 16 + lm;
  int bidx0 = (col / HW) * sB + (col % HW);
  v8f acc = {};
  for (int k0 = 0; k0 < K; k0 += 32) {
    v16bf a, b;
#pragma unroll
    for (int j = 0; j < 16; ++j) {
      a[j] = (__bf16)W[(mt * 16 + lm) * K + k0 + amap(j, half)];
      b[j] = (__bf16)X[bidx0 + (k0 + amap(j, half)) * sC];
    }
    acc = WMMA_BF16(a, b, acc);
  }
#pragma unroll
  for (int r = 0; r < 8; ++r) {
    int row = mt * 16 + r + half * 8;
    Y[(size_t)row * N + col] = acc[r];
  }
}

// cbn + prelu, channel-major [C][N], in->out same layout
__global__ void k_cbn_prelu(const float* __restrict__ y, float* __restrict__ o,
                            const float* __restrict__ acc, const float* __restrict__ g,
                            const float* __restrict__ b, const float* __restrict__ sb,
                            const float* __restrict__ pre, int C, int N) {
  long i = blockIdx.x * (long)blockDim.x + threadIdx.x;
  if (i >= (long)C * N) return;
  int c = (int)(i / N);
  float inv = 1.f / (float)N;
  float mu = acc[c] * inv, va = acc[C + c] * inv - mu * mu;
  float v = (y[i] - mu) * rsqrtf(va + EPS) * g[c] + b[c];
  v = sb[c] * v + sb[C + c];
  float a = pre[0];
  o[i] = v >= 0.f ? v : a * v;
}

// ---------------- axial attention kernels ----------------
// qkv[oc][col] = sum_c W[oc][c] * x1(c, b, spatial);  col = b*16384 + d1*128 + a
__global__ void k_qkv(const float* __restrict__ W, const float* __restrict__ x1,
                      float* __restrict__ qkv, int dim) {
  int lane = threadIdx.x & 31, wv = threadIdx.x >> 5;
  int half = lane >> 4, lm = lane & 15;
  int mt = blockIdx.y;                 // 0..7 (oc tiles)
  int nt = blockIdx.x * 4 + wv;        // 0..4095
  int col = nt * 16 + lm;
  int bb = col >> 14, d1 = (col >> 7) & 127, a = col & 127;
  int sp = bb * 16384 + (dim == 0 ? a * 128 + d1 : d1 * 128 + a);
  v8f acc = {};
  for (int k0 = 0; k0 < 64; k0 += 32) {
    v16bf av, bv;
#pragma unroll
    for (int j = 0; j < 16; ++j) {
      av[j] = (__bf16)W[(mt * 16 + lm) * 64 + k0 + amap(j, half)];
      bv[j] = (__bf16)x1[(k0 + amap(j, half)) * 65536 + sp];
    }
    acc = WMMA_BF16(av, bv, acc);
  }
#pragma unroll
  for (int r = 0; r < 8; ++r) {
    int row = mt * 16 + r + half * 8;
    qkv[(size_t)row * 65536 + col] = acc[r];
  }
}

// q_emb/k_emb: for fixed i, [(n,g), j] = sum_{c<8} qhat[n,g,c,i] * rel[c][i-j+127] (K padded 8->32)
__global__ void k_qemb(const float* __restrict__ qkv, const float* __restrict__ accS,
                       const float* __restrict__ gg, const float* __restrict__ bb,
                       const float* __restrict__ rel, float* __restrict__ qe,
                       float* __restrict__ ke) {
  int lane = threadIdx.x & 31, wv = threadIdx.x >> 5;
  int half = lane >> 4, lm = lane & 15;
  int i  = blockIdx.z;                  // 0..127
  int mt = blockIdx.y;                  // 0..127
  int jt = blockIdx.x * 4 + wv;         // 0..7
  int m = mt * 16 + lm;                 // (n,g) pair
  int n = m >> 2, g = m & 3;
  const float Ninv = 1.f / 65536.f;
  v16bf aq, ak, bq, bk;
#pragma unroll
  for (int j = 0; j < 16; ++j) {
    int k = amap(j, half);
    float qv = 0.f, kv = 0.f;
    if (k < 8) {
      int ocq = g * 32 + k, ock = g * 32 + 8 + k;
      int cidx = n * 128 + i;
      float mq = accS[ocq] * Ninv, vq = accS[128 + ocq] * Ninv - mq * mq;
      qv = (qkv[(size_t)ocq * 65536 + cidx] - mq) * rsqrtf(vq + EPS) * gg[ocq] + bb[ocq];
      float mk = accS[ock] * Ninv, vk = accS[128 + ock] * Ninv - mk * mk;
      kv = (qkv[(size_t)ock * 65536 + cidx] - mk) * rsqrtf(vk + EPS) * gg[ock] + bb[ock];
    }
    aq[j] = (__bf16)qv; ak[j] = (__bf16)kv;
    int kb = amap(j, half);
    float qb = 0.f, kbv = 0.f;
    if (kb < 8) {
      int jj = jt * 16 + lm;
      int ri = i - jj + 127;
      qb  = rel[kb * 255 + ri];
      kbv = rel[(8 + kb) * 255 + ri];
    }
    bq[j] = (__bf16)qb; bk[j] = (__bf16)kbv;
  }
  v8f a0 = {}, a1 = {};
  a0 = WMMA_BF16(aq, bq, a0);
  a1 = WMMA_BF16(ak, bk, a1);
#pragma unroll
  for (int r = 0; r < 8; ++r) {
    int row = mt * 16 + r + half * 8;
    int jj = jt * 16 + lm;
    qe[(size_t)row * 16384 + i * 128 + jj] = a0[r];
    ke[(size_t)row * 16384 + i * 128 + jj] = a1[r];
  }
}

// qk[n,g,i,j] = sum_p qe[n,g,p,i] * ke[n,g,p,j]  (batched 128x128x128)
// K-panels staged through LDS with GLOBAL_LOAD_ASYNC_TO_LDS_B128 (ASYNCcnt).
__global__ void __launch_bounds__(128) k_qk(const float* __restrict__ qe,
                                            const float* __restrict__ ke,
                                            float* __restrict__ qk) {
  __shared__ __align__(16) float shQ[32 * 128];
  __shared__ __align__(16) float shK[32 * 128];
  int tid = threadIdx.x;
  int lane = tid & 31, wv = tid >> 5;
  int half = lane >> 4, lm = lane & 15;
  int ng = blockIdx.y;
  int t = blockIdx.x * 4 + wv;          // 0..63
  int it = t >> 3, jt = t & 7;
  const float* Q = qe + (size_t)ng * 16384;
  const float* Kp = ke + (size_t)ng * 16384;
  v8f acc = {};
  for (int k0 = 0; k0 < 128; k0 += 32) {
    if (k0) __syncthreads();            // LDS reuse across K-steps
    async_copy_panel(Q + k0 * 128, shQ, tid);
    async_copy_panel(Kp + k0 * 128, shK, tid);
    wait_async0();                      // each wave waits for its own async loads
    __syncthreads();                    // publish all 4 waves' panels
    v16bf a, b;
#pragma unroll
    for (int j = 0; j < 16; ++j) {
      a[j] = (__bf16)shQ[amap(j, half) * 128 + it * 16 + lm];
      b[j] = (__bf16)shK[amap(j, half) * 128 + jt * 16 + lm];
    }
    acc = WMMA_BF16(a, b, acc);
  }
#pragma unroll
  for (int r = 0; r < 8; ++r) {
    int row = it * 16 + r + half * 8;
    qk[(size_t)ng * 16384 + row * 128 + jt * 16 + lm] = acc[r];
  }
}

// sim = softmax over j of (bn(qk)+bn(qe)+bn(ke)); one wave per (n,g,i) row; in-place into qk
__global__ void k_softmax(float* __restrict__ qk, const float* __restrict__ qe,
                          const float* __restrict__ ke, const float* __restrict__ acc12,
                          const float* __restrict__ sg, const float* __restrict__ sb) {
  int wv = threadIdx.x >> 5, lane = threadIdx.x & 31;
  long row = (long)blockIdx.x * 4 + wv;   // = ng*128 + i
  long ng = row >> 7;
  int g = (int)(ng & 3);
  const float Ninv = 1.f / 8388608.f;
  float sc[3], bi[3];
#pragma unroll
  for (int r = 0; r < 3; ++r) {
    int ch = r * 4 + g;
    float mu = acc12[ch] * Ninv, va = acc12[12 + ch] * Ninv - mu * mu;
    float rs = rsqrtf(va + EPS) * sg[ch];
    sc[r] = rs; bi[r] = sb[ch] - mu * rs;
  }
  size_t base = (size_t)row * 128;
  float vals[4]; float mx = -3.0e38f;
#pragma unroll
  for (int u = 0; u < 4; ++u) {
    int j = lane + u * 32;
    float v = qk[base + j] * sc[0] + bi[0] + qe[base + j] * sc[1] + bi[1] +
              ke[base + j] * sc[2] + bi[2];
    vals[u] = v; mx = fmaxf(mx, v);
  }
  for (int off = 16; off > 0; off >>= 1) mx = fmaxf(mx, __shfl_xor(mx, off, 32));
  float s = 0.f;
#pragma unroll
  for (int u = 0; u < 4; ++u) { vals[u] = __expf(vals[u] - mx); s += vals[u]; }
  for (int off = 16; off > 0; off >>= 1) s += __shfl_xor(s, off, 32);
  float inv = 1.f / s;
#pragma unroll
  for (int u = 0; u < 4; ++u) qk[base + lane + u * 32] = vals[u] * inv;
}

// am[n,g,c,i] = sum_j vhat[n,g,c,j]*sim[n,g,i,j]   (M=16,N=128,K=128 per (n,g))
__global__ void k_am(const float* __restrict__ sim, const float* __restrict__ qkv,
                     const float* __restrict__ accS, const float* __restrict__ gg,
                     const float* __restrict__ bb, float* __restrict__ am) {
  int lane = threadIdx.x & 31, wv = threadIdx.x >> 5;
  int half = lane >> 4, lm = lane & 15;
  int ng = blockIdx.y; int n = ng >> 2, g = ng & 3;
  int it = blockIdx.x * 4 + wv;          // 0..7
  int oc = g * 32 + 16 + lm;
  const float Ninv = 1.f / 65536.f;
  float mu = accS[oc] * Ninv, va = accS[128 + oc] * Ninv - mu * mu;
  float sc = rsqrtf(va + EPS) * gg[oc];
  float bi = bb[oc] - mu * sc;
  v8f acc = {};
  for (int k0 = 0; k0 < 128; k0 += 32) {
    v16bf a, b;
#pragma unroll
    for (int j = 0; j < 16; ++j) {
      a[j] = (__bf16)(qkv[(size_t)oc * 65536 + n * 128 + k0 + amap(j, half)] * sc + bi);
      b[j] = (__bf16)sim[(size_t)ng * 16384 + (it * 16 + lm) * 128 + k0 + amap(j, half)];
    }
    acc = WMMA_BF16(a, b, acc);
  }
#pragma unroll
  for (int r = 0; r < 8; ++r) {
    int c = r + half * 8;
    am[(size_t)ng * 2048 + c * 128 + it * 16 + lm] = acc[r];
  }
}

// ame[n,g,c,i] = sum_j sim[n,g,i,j]*ve[c,i,j]; per i: M=(n,g)=2048, N=c=16, K=128
__global__ void k_ame(const float* __restrict__ sim, const float* __restrict__ rel,
                      float* __restrict__ ame) {
  int lane = threadIdx.x & 31, wv = threadIdx.x >> 5;
  int half = lane >> 4, lm = lane & 15;
  int i = blockIdx.y;
  int mt = blockIdx.x * 4 + wv;          // 0..127
  v8f acc = {};
  for (int k0 = 0; k0 < 128; k0 += 32) {
    v16bf a, b;
#pragma unroll
    for (int j = 0; j < 16; ++j) {
      a[j] = (__bf16)sim[(size_t)(mt * 16 + lm) * 16384 + i * 128 + k0 + amap(j, half)];
      int kb = k0 + amap(j, half);
      b[j] = (__bf16)rel[(16 + lm) * 255 + (i - kb + 127)];
    }
    acc = WMMA_BF16(a, b, acc);
  }
#pragma unroll
  for (int r = 0; r < 8; ++r) {
    int m = mt * 16 + r + half * 8;      // (n,g)
    ame[((size_t)m * 16 + lm) * 128 + i] = acc[r];
  }
}

// bn(am)+bn(ame) summed, scattered back into x1 [cc][b][h][w]
__global__ void k_outcmb(const float* __restrict__ am, const float* __restrict__ ame,
                         const float* __restrict__ accA, const float* __restrict__ accE,
                         const float* __restrict__ og, const float* __restrict__ ob,
                         float* __restrict__ x1, int dim) {
  long idx = blockIdx.x * (long)blockDim.x + threadIdx.x;  // 512*64*128
  if (idx >= 512L * 64 * 128) return;
  int a = (int)(idx & 127);
  long t = idx >> 7; int cc = (int)(t & 63); int n = (int)(t >> 6);
  int g = cc >> 4, c = cc & 15;
  float vA = am[(size_t)n * 8192 + g * 2048 + c * 128 + a];
  float vE = ame[(size_t)n * 8192 + g * 2048 + c * 128 + a];
  int ch = g * 16 + c;
  const float Ninv = 1.f / 65536.f;
  float muA = accA[ch] * Ninv, vaA = accA[64 + ch] * Ninv - muA * muA;
  float muE = accE[ch] * Ninv, vaE = accE[64 + ch] * Ninv - muE * muE;
  int ocA = g * 32 + c * 2, ocE = ocA + 1;
  float rA = rsqrtf(vaA + EPS) * og[ocA]; float bA = ob[ocA] - muA * rA;
  float rE = rsqrtf(vaE + EPS) * og[ocE]; float bE = ob[ocE] - muE * rE;
  float v = vA * rA + bA + vE * rE + bE;
  int bb = n >> 7, d1 = n & 127;
  int sp = bb * 16384 + (dim == 0 ? a * 128 + d1 : d1 * 128 + a);
  x1[(size_t)cc * 65536 + sp] = v;
}

// avg-pool 2x2: x1 [c][b][128][128] -> xp [c][b][64][64]
__global__ void k_pool(const float* __restrict__ x1, float* __restrict__ xp) {
  int idx = blockIdx.x * blockDim.x + threadIdx.x;  // 1,048,576
  if (idx >= 64 * 4 * 64 * 64) return;
  int w2 = idx & 63, h2 = (idx >> 6) & 63, b = (idx >> 12) & 3, c = idx >> 14;
  const float* p = x1 + (size_t)c * 65536 + b * 16384;
  int o = (2 * h2) * 128 + 2 * w2;
  xp[idx] = 0.25f * (p[o] + p[o + 1] + p[o + 128] + p[o + 129]);
}

// final: cbn2(y2) + avgpool2(input), prelu2 -> out (4,64,64,64)
__global__ void k_final(const float* __restrict__ y2, const float* __restrict__ inp,
                        const float* __restrict__ acc2, const float* __restrict__ g2,
                        const float* __restrict__ b2, const float* __restrict__ sb2,
                        const float* __restrict__ pre, float* __restrict__ out) {
  int idx = blockIdx.x * blockDim.x + threadIdx.x;  // 1,048,576
  if (idx >= 4 * 64 * 64 * 64) return;
  int w2 = idx & 63, h2 = (idx >> 6) & 63, o = (idx >> 12) & 63, b = idx >> 18;
  float y = y2[(size_t)o * 16384 + b * 4096 + h2 * 64 + w2];
  const float Ninv = 1.f / 16384.f;
  float mu = acc2[o] * Ninv, va = acc2[64 + o] * Ninv - mu * mu;
  float v = (y - mu) * rsqrtf(va + EPS) * g2[o] + b2[o];
  v = sb2[o] * v + sb2[64 + o];
  const float* ip = inp + ((size_t)(b * 64 + o)) * 16384;
  int off = (2 * h2) * 128 + 2 * w2;
  v += 0.25f * (ip[off] + ip[off + 1] + ip[off + 128] + ip[off + 129]);
  float a = pre[0];
  out[idx] = v >= 0.f ? v : a * v;
}

// ---------------- host orchestration ----------------
enum : size_t {
  OFF_Y1   = 0,
  OFF_X1   = OFF_Y1  + 4194304,
  OFF_QKV  = OFF_X1  + 4194304,
  OFF_QE   = OFF_QKV + 8388608,
  OFF_KE   = OFF_QE  + 33554432,
  OFF_QK   = OFF_KE  + 33554432,
  OFF_AM   = OFF_QK  + 33554432,
  OFF_AME  = OFF_AM  + 4194304,
  OFF_XP   = OFF_AME + 4194304,
  OFF_Y2   = OFF_XP  + 1048576,
  OFF_SB1  = OFF_Y2  + 1048576,
  OFF_SB2  = OFF_SB1 + 128,
  OFF_ACC1 = OFF_SB2 + 128,
  OFF_ACCQ = OFF_ACC1 + 128,
  OFF_ACC12= OFF_ACCQ + 256,
  OFF_ACCO = OFF_ACC12 + 24,
  OFF_ACC2 = OFF_ACCO + 256,
};

static void run_axial(float* ws, int dim,
                      const float* qkv_w, const float* qkv_g, const float* qkv_b,
                      const float* sim_g, const float* sim_b, const float* rel,
                      const float* out_g, const float* out_b, hipStream_t stream) {
  float* x1   = ws + OFF_X1;
  float* qkv  = ws + OFF_QKV;
  float* qe   = ws + OFF_QE;
  float* ke   = ws + OFF_KE;
  float* qk   = ws + OFF_QK;
  float* am   = ws + OFF_AM;
  float* ame  = ws + OFF_AME;
  float* accQ = ws + OFF_ACCQ;
  float* ac12 = ws + OFF_ACC12;
  float* accO = ws + OFF_ACCO;

  k_qkv<<<dim3(1024, 8), 128, 0, stream>>>(qkv_w, x1, qkv, dim);
  k_zero<<<1, 256, 0, stream>>>(accQ, 256);
  k_stats<<<dim3(32, 128), 256, 0, stream>>>(qkv, accQ, 128, 65536);
  k_qemb<<<dim3(2, 128, 128), 128, 0, stream>>>(qkv, accQ, qkv_g, qkv_b, rel, qe, ke);
  k_qk<<<dim3(16, 2048), 128, 0, stream>>>(qe, ke, qk);
  k_zero<<<1, 64, 0, stream>>>(ac12, 24);
  k_stats_ng<<<dim3(256, 4), 256, 0, stream>>>(qk, ac12, 0);
  k_stats_ng<<<dim3(256, 4), 256, 0, stream>>>(qe, ac12, 4);
  k_stats_ng<<<dim3(256, 4), 256, 0, stream>>>(ke, ac12, 8);
  k_softmax<<<65536, 128, 0, stream>>>(qk, qe, ke, ac12, sim_g, sim_b);
  k_am<<<dim3(2, 2048), 128, 0, stream>>>(qk, qkv, accQ, qkv_g, qkv_b, am);
  k_ame<<<dim3(32, 128), 128, 0, stream>>>(qk, rel, ame);
  k_zero<<<1, 256, 0, stream>>>(accO, 256);
  k_stats_mid<<<dim3(32, 64), 256, 0, stream>>>(am, accO);
  k_stats_mid<<<dim3(32, 64), 256, 0, stream>>>(ame, accO + 128);
  k_outcmb<<<16384, 256, 0, stream>>>(am, ame, accO, accO + 128, out_g, out_b, x1, dim);
}

extern "C" void kernel_launch(void* const* d_in, const int* in_sizes, int n_in,
                              void* d_out, int out_size, void* d_ws, size_t ws_size,
                              hipStream_t stream) {
  const float* input    = (const float*)d_in[0];
  const float* latent   = (const float*)d_in[1];
  const float* w_in     = (const float*)d_in[2];
  const float* cbn1_g   = (const float*)d_in[3];
  const float* cbn1_b   = (const float*)d_in[4];
  const float* cbn1_lin = (const float*)d_in[5];
  const float* prelu1   = (const float*)d_in[6];
  const float* w_out    = (const float*)d_in[23];
  const float* cbn2_g   = (const float*)d_in[24];
  const float* cbn2_b   = (const float*)d_in[25];
  const float* cbn2_lin = (const float*)d_in[26];
  const float* prelu2   = (const float*)d_in[27];

  float* ws  = (float*)d_ws;
  float* y1  = ws + OFF_Y1;
  float* x1  = ws + OFF_X1;
  float* xp  = ws + OFF_XP;
  float* y2  = ws + OFF_Y2;
  float* sb1 = ws + OFF_SB1;
  float* sb2 = ws + OFF_SB2;
  float* ac1 = ws + OFF_ACC1;
  float* ac2 = ws + OFF_ACC2;

  k_sb<<<1, 128, 0, stream>>>(cbn1_lin, latent, sb1, 128, 256);
  k_sb<<<1, 128, 0, stream>>>(cbn2_lin, latent, sb2, 128, 256);

  // conv_in: y1[64][65536], input is (b,c,h,w)
  k_conv<<<dim3(1024, 4), 128, 0, stream>>>(w_in, input, y1, 65536, 16384, 64 * 16384, 16384);
  k_zero<<<1, 128, 0, stream>>>(ac1, 128);
  k_stats<<<dim3(32, 64), 256, 0, stream>>>(y1, ac1, 64, 65536);
  k_cbn_prelu<<<16384, 256, 0, stream>>>(y1, x1, ac1, cbn1_g, cbn1_b, sb1, prelu1, 64, 65536);

  run_axial(ws, 0, (const float*)d_in[7], (const float*)d_in[8], (const float*)d_in[9],
            (const float*)d_in[10], (const float*)d_in[11], (const float*)d_in[12],
            (const float*)d_in[13], (const float*)d_in[14], stream);
  run_axial(ws, 1, (const float*)d_in[15], (const float*)d_in[16], (const float*)d_in[17],
            (const float*)d_in[18], (const float*)d_in[19], (const float*)d_in[20],
            (const float*)d_in[21], (const float*)d_in[22], stream);

  k_pool<<<4096, 256, 0, stream>>>(x1, xp);
  // conv_out: y2[64][16384], xp is [c][b][64][64]
  k_conv<<<dim3(256, 4), 128, 0, stream>>>(w_out, xp, y2, 16384, 4096, 4096, 16384);
  k_zero<<<1, 128, 0, stream>>>(ac2, 128);
  k_stats<<<dim3(8, 64), 256, 0, stream>>>(y2, ac2, 64, 16384);
  k_final<<<4096, 256, 0, stream>>>(y2, input, ac2, cbn2_g, cbn2_b, sb2, prelu2,
                                    (float*)d_out);
}
